// Pts_Encoder_curvature_25426206392799
// MI455X (gfx1250) — matrix operand order repaired and checked by reading, not compiled, not run on hardware
//
#include <hip/hip_runtime.h>
#include <hip/hip_bf16.h>
#include <cstdint>
#include <cstddef>

// ---------------------------------------------------------------------------
// PointNet++ MSG encoder for MI455X (gfx1250, wave32).
//  - FPS: one workgroup/batch, dist[] in LDS, block argmax per iteration.
//  - Ball query: thread/query sequential scan (first-K-in-ball == reference).
//  - MLP layers: f16-in / f32-accum WMMA GEMM (v_wmma_f32_16x16x32_f16),
//    fragments per CDNA5 ISA 7.12.2 layouts, bias+ReLU fused epilogue.
//  - Max-over-K reduction writes transposed (B,C,S) outputs directly.
// Workspace requirement: ~91 MB (carved below).
// ---------------------------------------------------------------------------

typedef __attribute__((ext_vector_type(16))) _Float16 v16h;
typedef __attribute__((ext_vector_type(8)))  _Float16 v8h;
typedef __attribute__((ext_vector_type(8)))  float    v8f;

#define BATCH 8
#define NPTS  4096

static inline int cdiv_h(int a, int b){ return (a + b - 1) / b; }
static inline int pad32i(int x){ return (x + 31) & ~31; }
static inline size_t alignup(size_t x){ return (x + 255) & ~(size_t)255; }

// ------------------------------ kernels ------------------------------------

// (B,3,N) -> (B,N,3)
__global__ void k_transpose_l0(const float* __restrict__ xyz, float* __restrict__ l0)
{
    int idx = blockIdx.x * blockDim.x + threadIdx.x;
    int total = BATCH * 3 * NPTS;
    if (idx >= total) return;
    int b   = idx / (3 * NPTS);
    int rem = idx % (3 * NPTS);
    int c   = rem / NPTS;
    int n   = rem % NPTS;
    l0[((size_t)b * NPTS + n) * 3 + c] = xyz[idx];
}

// Farthest point sampling. grid = BATCH blocks, 1024 threads (32 waves/WGP).
__global__ void k_fps(const float* __restrict__ xyz, int n, int S,
                      int* __restrict__ fidx, float* __restrict__ nxyz)
{
    __shared__ float dist[4096];
    __shared__ float rv[1024];
    __shared__ int   ri[1024];
    __shared__ int   far_s;

    int b = blockIdx.x;
    int t = threadIdx.x;
    const float* P = xyz + (size_t)b * n * 3;

    for (int i = t; i < n; i += blockDim.x) dist[i] = 1e10f;
    if (t == 0) far_s = 0;
    __syncthreads();

    for (int it = 0; it < S; ++it) {
        int f = far_s;
        float cx = P[f * 3 + 0], cy = P[f * 3 + 1], cz = P[f * 3 + 2];
        if (t == 0) {
            fidx[b * S + it] = f;
            nxyz[((size_t)b * S + it) * 3 + 0] = cx;
            nxyz[((size_t)b * S + it) * 3 + 1] = cy;
            nxyz[((size_t)b * S + it) * 3 + 2] = cz;
        }
        float best = -1.0f; int bi = 0;
        for (int i = t; i < n; i += blockDim.x) {
            float dx = P[i * 3 + 0] - cx;
            float dy = P[i * 3 + 1] - cy;
            float dz = P[i * 3 + 2] - cz;
            float d  = dx * dx + dy * dy + dz * dz;
            float dd = fminf(dist[i], d);
            dist[i] = dd;
            if (dd > best) { best = dd; bi = i; }
        }
        rv[t] = best; ri[t] = bi;
        __syncthreads();
        for (int off = blockDim.x >> 1; off > 0; off >>= 1) {
            if (t < off && rv[t + off] > rv[t]) { rv[t] = rv[t + off]; ri[t] = ri[t + off]; }
            __syncthreads();
        }
        if (t == 0) far_s = ri[0];
        __syncthreads();
    }
}

// Ball query: first K indices (in index order) with d^2 <= r^2; pad with first.
__global__ void k_ball(const float* __restrict__ xyz, const float* __restrict__ nxyz,
                       int* __restrict__ gidx, int n, int S, int K, float r2)
{
    int idx = blockIdx.x * blockDim.x + threadIdx.x;
    if (idx >= BATCH * S) return;
    int b = idx / S;
    const float* Q = nxyz + (size_t)idx * 3;
    float qx = Q[0], qy = Q[1], qz = Q[2];
    const float* P = xyz + (size_t)b * n * 3;
    int* out = gidx + (size_t)idx * K;
    int cnt = 0, first = -1;
    for (int i = 0; i < n && cnt < K; ++i) {
        float dx = P[i * 3 + 0] - qx;
        float dy = P[i * 3 + 1] - qy;
        float dz = P[i * 3 + 2] - qz;
        float d2 = dx * dx + dy * dy + dz * dz;
        if (d2 <= r2) { if (first < 0) first = i; out[cnt++] = i; }
    }
    if (first < 0) first = 0;
    for (; cnt < K; ++cnt) out[cnt] = first;
}

// Gather + concat features into padded f16 activation matrix X (rows x cinpad).
// mode_sa==0 -> [points(Cp), xyz-diff(3)]; mode_sa==1 -> [xyz-diff(3), points(Cp)].
__global__ void k_gather(const float* __restrict__ xyz, const float* __restrict__ nxyz,
                         const float* __restrict__ pts, const int* __restrict__ gidx,
                         int n, int S, int K, int Cp, int cinpad, int mode_sa,
                         _Float16* __restrict__ X, int b, int s0, int s_chunk)
{
    int idx = blockIdx.x * blockDim.x + threadIdx.x;
    int total = s_chunk * K * cinpad;
    if (idx >= total) return;
    int c = idx % cinpad;
    int r = idx / cinpad;
    int j = r % K;
    int s = r / K;
    int gs = s0 + s;
    int g  = gidx[((size_t)b * S + gs) * K + j];
    float val = 0.0f;
    if (!mode_sa) {
        if (c < Cp) {
            val = pts[((size_t)b * n + g) * Cp + c];
        } else if (c < Cp + 3) {
            int d = c - Cp;
            val = xyz[((size_t)b * n + g) * 3 + d] - nxyz[((size_t)b * S + gs) * 3 + d];
        }
    } else {
        if (c < 3) {
            val = xyz[((size_t)b * n + g) * 3 + c] - nxyz[((size_t)b * S + gs) * 3 + c];
        } else if (c < 3 + Cp) {
            val = pts[((size_t)b * n + g) * Cp + (c - 3)];
        }
    }
    X[idx] = (_Float16)val;
}

// fp32 weights/bias -> zero-padded f16 weights (kp x np) + f32 bias (np).
__global__ void k_wconv(const float* __restrict__ W, const float* __restrict__ bsrc,
                        _Float16* __restrict__ Wh, float* __restrict__ bh,
                        int cin, int cout, int kp, int np)
{
    int idx = blockIdx.x * blockDim.x + threadIdx.x;
    if (idx < np) bh[idx] = (idx < cout) ? bsrc[idx] : 0.0f;
    if (idx >= kp * np) return;
    int i = idx / np, j = idx % np;
    Wh[idx] = (i < cin && j < cout) ? (_Float16)W[(size_t)i * cout + j] : (_Float16)0.0f;
}

// Tiled WMMA GEMM: Y = relu(X[rows x kp] * W[kp x np] + bias).
// One wave -> one 16x16 C tile, 8 waves per block. Fragment layouts per
// CDNA5 ISA 7.12.2: A lane(l): row m = l&15, halves = K[kb..kb+7],K[16+kb..],
// kb = (l>>4)*8; B lane(l): col N = l&15, same K order as A.
__global__ void k_gemm_wmma(const _Float16* __restrict__ X, const _Float16* __restrict__ W,
                            const float* __restrict__ bias,
                            _Float16* __restrict__ Yh, float* __restrict__ Yf,
                            int rows, int kp, int np)
{
    int nt_n  = np >> 4;
    int mt_n  = rows >> 4;
    int wave  = blockIdx.x * (blockDim.x >> 5) + (threadIdx.x >> 5);
    int lane  = threadIdx.x & 31;
    if (wave >= mt_n * nt_n) return;           // wave-uniform: EXEC stays all-1s
    int mt = wave / nt_n, nt = wave % nt_n;

    int m  = mt * 16 + (lane & 15);
    int kb = (lane >> 4) * 8;
    const _Float16* xrow = X + (size_t)m * kp;

    v8f acc = {};
    for (int k0 = 0; k0 < kp; k0 += 32) {
        v8h lo = *(const v8h*)(xrow + k0 + kb);
        v8h hi = *(const v8h*)(xrow + k0 + 16 + kb);
        v16h a;
#pragma unroll
        for (int i = 0; i < 8; ++i) { a[i] = lo[i]; a[8 + i] = hi[i]; }
        v16h bf;
        int nb = nt * 16 + (lane & 15);
#pragma unroll
        for (int i = 0; i < 8; ++i) { bf[i] = W[(size_t)(k0 + kb + i) * np + nb]; bf[8 + i] = W[(size_t)(k0 + 16 + kb + i) * np + nb]; }
        acc = __builtin_amdgcn_wmma_f32_16x16x32_f16(false, a, false, bf,
                                                     (short)0, acc, false, false);
    }

    int n     = nt * 16 + (lane & 15);
    int mbase = mt * 16 + (lane >> 4) * 8;
    float bv  = bias[n];
#pragma unroll
    for (int v = 0; v < 8; ++v) {
        float y = acc[v] + bv;
        y = y > 0.0f ? y : 0.0f;
        int mr = mbase + v;
        if (Yh) Yh[(size_t)mr * np + n] = (_Float16)y;
        else    Yf[(size_t)mr * np + n] = y;
    }
}

// Max over the K grouped points; write transposed (B,Ctot,S) output and
// optionally the untransposed (B,S,Ctot) feature buffer for the next stage.
__global__ void k_maxk(const float* __restrict__ Yf, int K, int s_chunk, int cout, int np,
                       float* __restrict__ out_t, float* __restrict__ pts_buf,
                       int S, int Ctot, int b, int s0, int c0)
{
    int idx = blockIdx.x * blockDim.x + threadIdx.x;
    if (idx >= s_chunk * cout) return;
    int s = idx / cout, c = idx % cout;
    const float* p = Yf + (size_t)(s * K) * np + c;
    float m = p[0];
    for (int j = 1; j < K; ++j) m = fmaxf(m, p[(size_t)j * np]);
    out_t[(size_t)b * Ctot * S + (size_t)(c0 + c) * S + (s0 + s)] = m;
    if (pts_buf) pts_buf[((size_t)b * S + (s0 + s)) * Ctot + (c0 + c)] = m;
}

// (B,S,3) -> (B,3,S)
__global__ void k_txyz(const float* __restrict__ nx, float* __restrict__ dst, int S)
{
    int idx = blockIdx.x * blockDim.x + threadIdx.x;
    if (idx >= BATCH * S * 3) return;
    int b   = idx / (S * 3);
    int rem = idx % (S * 3);
    int s   = rem / 3;
    int c   = rem % 3;
    dst[(size_t)b * 3 * S + (size_t)c * S + s] = nx[((size_t)b * S + s) * 3 + c];
}

// dst[b,s] = src[b, fidx[b,s]]
__global__ void k_curv(const float* __restrict__ src, const int* __restrict__ fidx,
                       float* __restrict__ dst, int S, int n)
{
    int idx = blockIdx.x * blockDim.x + threadIdx.x;
    if (idx >= BATCH * S) return;
    int b = idx / S;
    dst[idx] = src[(size_t)b * n + fidx[idx]];
}

// ------------------------------ host side ----------------------------------

struct LayerP { const float* W; const float* b; int cin; int cout; };

struct WS {
    float* l0; int* fidx;
    float *xyz1, *xyz2, *xyz3, *c1, *c2;
    int* gidx;
    _Float16 *X, *Y0, *Y1;
    float* Yf;
    _Float16* Wh[3];
    float* bh[3];
    float *pts1, *pts2;
};

static void launch_gemm(hipStream_t st, const _Float16* X, const _Float16* Wh,
                        const float* bh, _Float16* Yh, float* Yf,
                        int rows, int kp, int np)
{
    int tiles  = (rows / 16) * (np / 16);
    int blocks = cdiv_h(tiles, 8);
    k_gemm_wmma<<<blocks, 256, 0, st>>>(X, Wh, bh, Yh, Yf, rows, kp, np);
}

static void run_branch(hipStream_t st, WS& ws,
                       const float* xyz, const float* nxyz, const float* pts, int Cp,
                       int n, int S, int K, float radius, int mode_sa,
                       const LayerP L[3],
                       float* out_t, float* pts_out, int Ctot, int c0)
{
    k_ball<<<cdiv_h(BATCH * S, 256), 256, 0, st>>>(xyz, nxyz, ws.gidx, n, S, K,
                                                   radius * radius);
    int kp[3], np[3];
    for (int l = 0; l < 3; ++l) {
        kp[l] = pad32i(L[l].cin);
        np[l] = pad32i(L[l].cout);
        k_wconv<<<cdiv_h(kp[l] * np[l], 256), 256, 0, st>>>(L[l].W, L[l].b,
                                                            ws.Wh[l], ws.bh[l],
                                                            L[l].cin, L[l].cout,
                                                            kp[l], np[l]);
    }
    const int ROWS_MAX = 16384;
    int s_chunk = S < (ROWS_MAX / K) ? S : (ROWS_MAX / K);
    for (int b = 0; b < BATCH; ++b) {
        for (int s0 = 0; s0 < S; s0 += s_chunk) {
            int rows = s_chunk * K;
            k_gather<<<cdiv_h(rows * kp[0], 256), 256, 0, st>>>(
                xyz, nxyz, pts, ws.gidx, n, S, K, Cp, kp[0], mode_sa,
                ws.X, b, s0, s_chunk);
            launch_gemm(st, ws.X,  ws.Wh[0], ws.bh[0], ws.Y0, nullptr, rows, kp[0], np[0]);
            launch_gemm(st, ws.Y0, ws.Wh[1], ws.bh[1], ws.Y1, nullptr, rows, np[0], np[1]);
            launch_gemm(st, ws.Y1, ws.Wh[2], ws.bh[2], nullptr, ws.Yf, rows, np[1], np[2]);
            k_maxk<<<cdiv_h(s_chunk * L[2].cout, 256), 256, 0, st>>>(
                ws.Yf, K, s_chunk, L[2].cout, np[2],
                out_t, pts_out, S, Ctot, b, s0, c0);
        }
    }
}

// output element offsets (fp32) in d_out, per reference return tuple
#define OFF_XYZ0 0
#define OFF_L1X  98304
#define OFF_L2X  122880
#define OFF_L3X  129024
#define OFF_XYZ1 130560
#define OFF_L1P  228864
#define OFF_L2P  2850304
#define OFF_L3P  3898880
#define OFF_C3   4292096

extern "C" void kernel_launch(void* const* d_in, const int* in_sizes, int n_in,
                              void* d_out, int out_size, void* d_ws, size_t ws_size,
                              hipStream_t stream)
{
    const float* xyz  = (const float*)d_in[0];
    const float* curv = (const float*)d_in[1];
    float* out = (float*)d_out;

    // ---- carve workspace ----
    char* p = (char*)d_ws;
    auto take = [&](size_t bytes) -> void* { void* r = (void*)p; p += alignup(bytes); return r; };
    WS ws;
    ws.l0    = (float*)take((size_t)BATCH * NPTS * 3 * 4);
    ws.fidx  = (int*)  take((size_t)BATCH * 1024 * 4);
    ws.xyz1  = (float*)take((size_t)BATCH * 1024 * 3 * 4);
    ws.xyz2  = (float*)take((size_t)BATCH * 256 * 3 * 4);
    ws.xyz3  = (float*)take((size_t)BATCH * 64 * 3 * 4);
    ws.c1    = (float*)take((size_t)BATCH * 1024 * 4);
    ws.c2    = (float*)take((size_t)BATCH * 256 * 4);
    ws.gidx  = (int*)  take((size_t)BATCH * 1024 * 128 * 4);
    ws.X     = (_Float16*)take((size_t)16384 * 544 * 2);
    ws.Y0    = (_Float16*)take((size_t)16384 * 512 * 2);
    ws.Y1    = (_Float16*)take((size_t)16384 * 512 * 2);
    ws.Yf    = (float*)take((size_t)16384 * 256 * 4);
    for (int l = 0; l < 3; ++l) {
        ws.Wh[l] = (_Float16*)take((size_t)544 * 768 * 2);
        ws.bh[l] = (float*)take((size_t)768 * 4);
    }
    ws.pts1 = (float*)take((size_t)BATCH * 1024 * 320 * 4);
    ws.pts2 = (float*)take((size_t)BATCH * 256 * 512 * 4);

    auto layers = [&](int base, const int* cins, const int* couts, LayerP* L) {
        for (int l = 0; l < 3; ++l) {
            L[l].W = (const float*)d_in[base + 2 * l];
            L[l].b = (const float*)d_in[base + 2 * l + 1];
            L[l].cin = cins[l]; L[l].cout = couts[l];
        }
    };

    // pass-through xyz outputs
    hipMemcpyAsync(out + OFF_XYZ0, xyz, (size_t)BATCH * 3 * NPTS * 4,
                   hipMemcpyDeviceToDevice, stream);
    hipMemcpyAsync(out + OFF_XYZ1, xyz, (size_t)BATCH * 3 * NPTS * 4,
                   hipMemcpyDeviceToDevice, stream);

    k_transpose_l0<<<cdiv_h(BATCH * 3 * NPTS, 256), 256, 0, stream>>>(xyz, ws.l0);

    // ---------------- stage 1 (S=1024) ----------------
    k_fps<<<BATCH, 1024, 0, stream>>>(ws.l0, NPTS, 1024, ws.fidx, ws.xyz1);
    k_txyz<<<cdiv_h(BATCH * 1024 * 3, 256), 256, 0, stream>>>(ws.xyz1, out + OFF_L1X, 1024);
    k_curv<<<cdiv_h(BATCH * 1024, 256), 256, 0, stream>>>(curv, ws.fidx, ws.c1, 1024, NPTS);

    {
        LayerP L[3];
        int ci0[3] = {6, 32, 32},  co0[3] = {32, 32, 64};
        int ci1[3] = {6, 64, 64},  co1[3] = {64, 64, 128};
        int ci2[3] = {6, 64, 96},  co2[3] = {64, 96, 128};
        layers(2, ci0, co0, L);
        run_branch(stream, ws, ws.l0, ws.xyz1, ws.l0, 3, NPTS, 1024, 32, 0.1f, 0,
                   L, out + OFF_L1P, ws.pts1, 320, 0);
        layers(8, ci1, co1, L);
        run_branch(stream, ws, ws.l0, ws.xyz1, ws.l0, 3, NPTS, 1024, 64, 0.2f, 0,
                   L, out + OFF_L1P, ws.pts1, 320, 64);
        layers(14, ci2, co2, L);
        run_branch(stream, ws, ws.l0, ws.xyz1, ws.l0, 3, NPTS, 1024, 128, 0.4f, 0,
                   L, out + OFF_L1P, ws.pts1, 320, 192);
    }

    // ---------------- stage 2 (S=256) ----------------
    k_fps<<<BATCH, 1024, 0, stream>>>(ws.xyz1, 1024, 256, ws.fidx, ws.xyz2);
    k_txyz<<<cdiv_h(BATCH * 256 * 3, 256), 256, 0, stream>>>(ws.xyz2, out + OFF_L2X, 256);
    k_curv<<<cdiv_h(BATCH * 256, 256), 256, 0, stream>>>(ws.c1, ws.fidx, ws.c2, 256, 1024);

    {
        LayerP L[3];
        int ci0[3] = {323, 128, 128}, co0[3] = {128, 128, 256};
        int ci1[3] = {323, 128, 196}, co1[3] = {128, 196, 256};
        layers(20, ci0, co0, L);
        run_branch(stream, ws, ws.xyz1, ws.xyz2, ws.pts1, 320, 1024, 256, 64, 0.4f, 0,
                   L, out + OFF_L2P, ws.pts2, 512, 0);
        layers(26, ci1, co1, L);
        run_branch(stream, ws, ws.xyz1, ws.xyz2, ws.pts1, 320, 1024, 256, 128, 0.8f, 0,
                   L, out + OFF_L2P, ws.pts2, 512, 256);
    }

    // ---------------- stage 3 (S=64, single-radius SA) ----------------
    k_fps<<<BATCH, 1024, 0, stream>>>(ws.xyz2, 256, 64, ws.fidx, ws.xyz3);
    k_txyz<<<cdiv_h(BATCH * 64 * 3, 256), 256, 0, stream>>>(ws.xyz3, out + OFF_L3X, 64);
    k_curv<<<cdiv_h(BATCH * 64, 256), 256, 0, stream>>>(ws.c2, ws.fidx, out + OFF_C3, 64, 256);

    {
        LayerP L[3];
        int ci[3] = {515, 256, 512}, co[3] = {256, 512, 768};
        layers(32, ci, co, L);
        run_branch(stream, ws, ws.xyz2, ws.xyz3, ws.pts2, 512, 256, 64, 32, 0.6f, 1,
                   L, out + OFF_L3P, nullptr, 768, 0);
    }

    (void)in_sizes; (void)n_in; (void)out_size; (void)ws_size;
}
